// RPNHead_60705067762265
// MI455X (gfx1250) — matrix operand order repaired and checked by reading, not compiled, not run on hardware
//
#include <hip/hip_runtime.h>
#include <hip/hip_bf16.h>

// ---------------------------------------------------------------------------
// RPN head for MI455X (gfx1250, wave32, WMMA).
// Conv 3x3 (128->128, 256x256) as implicit GEMM via v_wmma_f32_16x16x32_bf16
// over a zero-padded bf16 activation buffer (branch-free inner loop, 64px x
// 16oc per wave = 4 WMMAs per B-fragment). Heads as a second WMMA GEMM, then
// decode / radix-select top-1000 / bitonic sort / greedy NMS / 300x5 output.
// ---------------------------------------------------------------------------

typedef __attribute__((ext_vector_type(16))) __bf16 v16bf;
typedef __attribute__((ext_vector_type(8)))  float  v8f;

union Frag {
    uint4 q[2];
    v16bf v;
};

#define NPIX   65536     // 256*256
#define CCH    128
#define PADW   258       // 256 + 1 halo each side
#define ROWSTR (PADW * CCH)   // padded row stride in elements
#define NANCH  589824    // NPIX*9
#define PRE_K  1000
#define POST_K 300
#define CANDN  2048

__device__ __forceinline__ unsigned short f2bf(float f) {
    unsigned u = __float_as_uint(f);
    unsigned r = u + 0x7FFFu + ((u >> 16) & 1u); // round-to-nearest-even
    return (unsigned short)(r >> 16);
}

__device__ __forceinline__ unsigned skey(float f) {
    unsigned u = __float_as_uint(f);
    return (u & 0x80000000u) ? ~u : (u | 0x80000000u);
}

// ---------------------------------------------------------------------------
// Pack conv weights [o][c][ky][kx] -> bf16 [tap][o][c]; head weights -> [o][c]
// (o: 0..8 cls, 9..44 reg, 45..47 zero pad).
// ---------------------------------------------------------------------------
__global__ void k_pack_w(const float* __restrict__ bw,
                         const float* __restrict__ cw,
                         const float* __restrict__ rw,
                         unsigned short* __restrict__ w9,
                         unsigned short* __restrict__ hw) {
    int i = blockIdx.x * 256 + threadIdx.x;
    const int NW = 9 * CCH * CCH;
    if (i < NW) {
        int tap = i / (CCH * CCH);
        int rem = i - tap * CCH * CCH;
        int o = rem / CCH, c = rem - o * CCH;
        w9[tap * (CCH * CCH) + o * CCH + c] = f2bf(bw[(o * CCH + c) * 9 + tap]);
    } else if (i < NW + 48 * CCH) {
        int j = i - NW;
        int o = j / CCH, c = j - o * CCH;
        float v = 0.f;
        if (o < 9)       v = cw[o * CCH + c];
        else if (o < 45) v = rw[(o - 9) * CCH + c];
        hw[j] = f2bf(v);
    }
}

// ---------------------------------------------------------------------------
// Zero the halo border of the padded activation buffer (rows 0/257, cols
// 0/257). Interior is fully overwritten by the transpose each call.
// 1028 border cells x 128 channels = 131584 elements -> 514 blocks x 256.
// ---------------------------------------------------------------------------
__global__ void k_zero_edge(unsigned short* __restrict__ ftp) {
    int i = blockIdx.x * 256 + threadIdx.x;
    int cell = i >> 7, c = i & 127;
    int row, col;
    if (cell < 258)      { row = 0;          col = cell; }
    else if (cell < 516) { row = 257;        col = cell - 258; }
    else if (cell < 772) { row = cell - 515; col = 0; }     // rows 1..256
    else                 { row = cell - 771; col = 257; }   // rows 1..256
    ftp[(size_t)(row * PADW + col) * CCH + c] = 0;
}

// ---------------------------------------------------------------------------
// Transpose fmap [c][pix] f32 -> padded bf16 [row+1][col+1][c] via LDS tile.
// ---------------------------------------------------------------------------
__global__ void k_tpose(const float* __restrict__ fmap,
                        unsigned short* __restrict__ ftp) {
    __shared__ float tile[32][33];
    int px0 = blockIdx.x * 32, c0 = blockIdx.y * 32;
    for (int i = 0; i < 4; ++i) {
        int c = threadIdx.y + i * 8;
        tile[c][threadIdx.x] = fmap[(size_t)(c0 + c) * NPIX + px0 + threadIdx.x];
    }
    __syncthreads();
    for (int i = 0; i < 4; ++i) {
        int p = threadIdx.y + i * 8;
        int pix = px0 + p;
        int y = pix >> 8, x = pix & 255;
        ftp[(size_t)((y + 1) * PADW + (x + 1)) * CCH + c0 + threadIdx.x] =
            f2bf(tile[threadIdx.x][p]);
    }
}

// ---------------------------------------------------------------------------
// Conv 3x3 + bias + relu as implicit GEMM (branch-free, padded input).
// Block = 256 thr = 8 waves; wave tile = 64 pixels x 16 out-channels
// (4 accumulators, B-fragment reused across 4 WMMAs; B shared block-wide).
// Grid: (65536/512, 128/16) = (128, 8). K loop: 9 taps x 4 steps of 32 ch.
// Output stored bf16 [pix][c] for the head GEMM.
// ---------------------------------------------------------------------------
__global__ void k_conv(const unsigned short* __restrict__ ftp,
                       const unsigned short* __restrict__ w9,
                       const float* __restrict__ bb,
                       unsigned short* __restrict__ xt) {
    int lane = threadIdx.x & 31, wv = threadIdx.x >> 5;
    int pw  = (blockIdx.x * 8 + wv) * 64;     // 64 consecutive px, same row
    int oc0 = blockIdx.y * 16;
    int m  = lane & 15, kh = lane >> 4;
    int y  = pw >> 8, x0 = pw & 255;
    int oc = oc0 + m;

    // per-subtile A base (element index into padded buffer), K-half applied
    size_t abase[4];
#pragma unroll
    for (int t = 0; t < 4; ++t) {
        int xg = x0 + t * 16 + m;             // global column of this pixel
        abase[t] = (size_t)((y + 1) * PADW + (xg + 1)) * CCH + kh * 8;
    }
    const unsigned short* wbase = w9 + oc * CCH;   // + tap*CCH*CCH later

    v8f acc0 = {0.f, 0.f, 0.f, 0.f, 0.f, 0.f, 0.f, 0.f};
    v8f acc1 = acc0, acc2 = acc0, acc3 = acc0;

    for (int tap = 0; tap < 9; ++tap) {
        int dy = tap / 3 - 1, dx = tap % 3 - 1;
        long toff = (long)(dy * PADW + dx) * CCH;
        const unsigned short* wtap = wbase + tap * (CCH * CCH);
        if (tap < 8) {   // warm next tap's A row + B tile in WGP cache
            int ndy = (tap + 1) / 3 - 1, ndx = (tap + 1) % 3 - 1;
            __builtin_prefetch(ftp + abase[0] + (long)(ndy * PADW + ndx) * CCH, 0, 3);
            __builtin_prefetch(wbase + (tap + 1) * (CCH * CCH), 0, 3);
        }
#pragma unroll
        for (int cs = 0; cs < 4; ++cs) {
            int c0 = cs * 32;
            Frag A0, A1, A2, A3, B;
            const uint4* p0 = reinterpret_cast<const uint4*>(ftp + abase[0] + toff + c0);
            const uint4* p1 = reinterpret_cast<const uint4*>(ftp + abase[1] + toff + c0);
            const uint4* p2 = reinterpret_cast<const uint4*>(ftp + abase[2] + toff + c0);
            const uint4* p3 = reinterpret_cast<const uint4*>(ftp + abase[3] + toff + c0);
            A0.q[0] = p0[0]; A0.q[1] = p0[2];   // K base+0..7 / base+16..23
            A1.q[0] = p1[0]; A1.q[1] = p1[2];
            A2.q[0] = p2[0]; A2.q[1] = p2[2];
            A3.q[0] = p3[0]; A3.q[1] = p3[2];
            const uint4* pb = reinterpret_cast<const uint4*>(wtap + c0 + kh * 8);
            B.q[0] = pb[0]; B.q[1] = pb[2];
            acc0 = __builtin_amdgcn_wmma_f32_16x16x32_bf16(false, A0.v, false, B.v, (short)0, acc0, false, false);
            acc1 = __builtin_amdgcn_wmma_f32_16x16x32_bf16(false, A1.v, false, B.v, (short)0, acc1, false, false);
            acc2 = __builtin_amdgcn_wmma_f32_16x16x32_bf16(false, A2.v, false, B.v, (short)0, acc2, false, false);
            acc3 = __builtin_amdgcn_wmma_f32_16x16x32_bf16(false, A3.v, false, B.v, (short)0, acc3, false, false);
        }
    }

    float bias = bb[oc];
    v8f accs[4] = {acc0, acc1, acc2, acc3};
#pragma unroll
    for (int t = 0; t < 4; ++t) {
#pragma unroll
        for (int r = 0; r < 8; ++r) {
            int pr = pw + t * 16 + r + 8 * kh;      // D layout: M = r + 8*kh
            float v = fmaxf(accs[t][r] + bias, 0.f);
            xt[(size_t)pr * CCH + oc] = f2bf(v);
        }
    }
}

// ---------------------------------------------------------------------------
// Head GEMM: out45[pix][48] = x[pix][c] * hw[o][c].
// Wave tile 64 px x 16 oc; grid (65536/512, 3); K = 128 in 4 steps.
// ---------------------------------------------------------------------------
__global__ void k_head(const unsigned short* __restrict__ xt,
                       const unsigned short* __restrict__ hw,
                       float* __restrict__ out45) {
    int lane = threadIdx.x & 31, wv = threadIdx.x >> 5;
    int pw  = (blockIdx.x * 8 + wv) * 64;
    int oc0 = blockIdx.y * 16;
    int m = lane & 15, kh = lane >> 4;
    int oc = oc0 + m;

    size_t abase[4];
#pragma unroll
    for (int t = 0; t < 4; ++t)
        abase[t] = (size_t)(pw + t * 16 + m) * CCH + kh * 8;
    const unsigned short* brow = hw + oc * CCH;

    v8f acc0 = {0.f, 0.f, 0.f, 0.f, 0.f, 0.f, 0.f, 0.f};
    v8f acc1 = acc0, acc2 = acc0, acc3 = acc0;
#pragma unroll
    for (int cs = 0; cs < 4; ++cs) {
        int c0 = cs * 32;
        Frag A0, A1, A2, A3, B;
        const uint4* p0 = reinterpret_cast<const uint4*>(xt + abase[0] + c0);
        const uint4* p1 = reinterpret_cast<const uint4*>(xt + abase[1] + c0);
        const uint4* p2 = reinterpret_cast<const uint4*>(xt + abase[2] + c0);
        const uint4* p3 = reinterpret_cast<const uint4*>(xt + abase[3] + c0);
        A0.q[0] = p0[0]; A0.q[1] = p0[2];
        A1.q[0] = p1[0]; A1.q[1] = p1[2];
        A2.q[0] = p2[0]; A2.q[1] = p2[2];
        A3.q[0] = p3[0]; A3.q[1] = p3[2];
        const uint4* pb = reinterpret_cast<const uint4*>(brow + c0 + kh * 8);
        B.q[0] = pb[0]; B.q[1] = pb[2];
        acc0 = __builtin_amdgcn_wmma_f32_16x16x32_bf16(false, A0.v, false, B.v, (short)0, acc0, false, false);
        acc1 = __builtin_amdgcn_wmma_f32_16x16x32_bf16(false, A1.v, false, B.v, (short)0, acc1, false, false);
        acc2 = __builtin_amdgcn_wmma_f32_16x16x32_bf16(false, A2.v, false, B.v, (short)0, acc2, false, false);
        acc3 = __builtin_amdgcn_wmma_f32_16x16x32_bf16(false, A3.v, false, B.v, (short)0, acc3, false, false);
    }
    v8f accs[4] = {acc0, acc1, acc2, acc3};
#pragma unroll
    for (int t = 0; t < 4; ++t)
#pragma unroll
        for (int r = 0; r < 8; ++r) {
            int pr = pw + t * 16 + r + 8 * kh;
            out45[(size_t)pr * 48 + oc] = accs[t][r];
        }
}

// ---------------------------------------------------------------------------
// Sigmoid scores + anchor decode (anchors generated on the fly).
// ---------------------------------------------------------------------------
__global__ void k_decode(const float* __restrict__ o45,
                         const float* __restrict__ clsb,
                         const float* __restrict__ regb,
                         float* __restrict__ scores,
                         float* __restrict__ boxes) {
    int i = blockIdx.x * 256 + threadIdx.x;   // grid sized exactly NANCH
    int pix = i / 9, a = i - pix * 9;
    int r = a / 3, s = a - r * 3;
    float ratio = (r == 0) ? 0.5f : ((r == 1) ? 1.f : 2.f);
    float size  = (s == 0) ? 128.f : ((s == 1) ? 256.f : 512.f);
    float hr = sqrtf(ratio);
    float wa = size / hr, ha = size * hr;
    float cx = (float)(pix & 255) * 4.f;
    float cy = (float)(pix >> 8) * 4.f;

    const float* row = o45 + (size_t)pix * 48;
    float logit = row[a] + clsb[a];
    scores[i] = 1.f / (1.f + expf(-logit));

    float d0 = row[9 + a * 4 + 0] + regb[a * 4 + 0];
    float d1 = row[9 + a * 4 + 1] + regb[a * 4 + 1];
    float d2 = row[9 + a * 4 + 2] + regb[a * 4 + 2];
    float d3 = row[9 + a * 4 + 3] + regb[a * 4 + 3];
    float ncx = d0 * wa + cx, ncy = d1 * ha + cy;
    float nw = wa * expf(d2), nh = ha * expf(d3);
    reinterpret_cast<float4*>(boxes)[i] =
        make_float4(ncx - nw * 0.5f, ncy - nh * 0.5f, ncx + nw * 0.5f, ncy + nh * 0.5f);
}

// --------------------------- top-1000 radix select -------------------------
__global__ void k_sel_init(unsigned* state) {
    state[0] = 0u;       // prefix
    state[1] = PRE_K;    // remaining k
    state[2] = 0u;       // compact counter
}

__global__ void k_zero_hist(unsigned* hist) { hist[threadIdx.x] = 0u; }

__global__ void k_zero_cand(unsigned* candk, unsigned* candi) {
    int i = blockIdx.x * 256 + threadIdx.x;
    if (i < CANDN) { candk[i] = 0u; candi[i] = 0x7FFFFFFFu; }
}

__global__ void k_hist(const float* __restrict__ scores,
                       const unsigned* __restrict__ state,
                       unsigned* __restrict__ hist, int shift, unsigned mask) {
    __shared__ unsigned lh[256];
    lh[threadIdx.x] = 0u;
    __syncthreads();
    int i = blockIdx.x * 256 + threadIdx.x;
    unsigned prefix = state[0];
    unsigned key = skey(scores[i]);
    if (((key ^ prefix) & mask) == 0u)
        atomicAdd(&lh[(key >> shift) & 255u], 1u);
    __syncthreads();
    atomicAdd(&hist[threadIdx.x], lh[threadIdx.x]);
}

__global__ void k_pick(unsigned* state, const unsigned* hist, int shift) {
    unsigned k = state[1], running = 0u, prefix = state[0];
    for (int d = 255; d >= 0; --d) {
        unsigned c = hist[d];
        if (running + c >= k) {
            state[0] = prefix | ((unsigned)d << shift);
            state[1] = k - running;
            return;
        }
        running += c;
    }
}

__global__ void k_compact(const float* __restrict__ scores,
                          unsigned* __restrict__ state,
                          unsigned* __restrict__ candk,
                          unsigned* __restrict__ candi) {
    int i = blockIdx.x * 256 + threadIdx.x;
    unsigned key = skey(scores[i]);
    if (key >= state[0]) {
        unsigned pos = atomicAdd(&state[2], 1u);
        if (pos < CANDN) { candk[pos] = key; candi[pos] = (unsigned)i; }
    }
}

// Bitonic sort of 2048 candidates: best (key desc, idx asc) first.
__global__ void k_sort(unsigned* candk, unsigned* candi) {
    __shared__ unsigned sk[CANDN], si[CANDN];
    int tid = threadIdx.x;
    for (int i = tid; i < CANDN; i += 1024) { sk[i] = candk[i]; si[i] = candi[i]; }
    __syncthreads();
    for (int k = 2; k <= CANDN; k <<= 1) {
        for (int j = k >> 1; j > 0; j >>= 1) {
            for (int i = tid; i < CANDN; i += 1024) {
                int l = i ^ j;
                if (l > i) {
                    bool up = ((i & k) == 0);
                    unsigned ka = sk[i], ia = si[i], kb = sk[l], ib = si[l];
                    bool l_better = (kb > ka) || (kb == ka && ib < ia);
                    bool sw = up ? l_better : !l_better;
                    if (sw) { sk[i] = kb; si[i] = ib; sk[l] = ka; si[l] = ia; }
                }
            }
            __syncthreads();
        }
    }
    for (int i = tid; i < CANDN; i += 1024) { candk[i] = sk[i]; candi[i] = si[i]; }
}

__global__ void k_gather(const unsigned* __restrict__ candi,
                         const float* __restrict__ scores,
                         const float* __restrict__ boxes,
                         float* __restrict__ selb,
                         float* __restrict__ sels) {
    int i = blockIdx.x * 256 + threadIdx.x;
    if (i < PRE_K) {
        unsigned idx = candi[i];
        float4 b = reinterpret_cast<const float4*>(boxes)[idx];
        selb[i * 4 + 0] = fminf(fmaxf(b.x, 0.f), 1024.f);
        selb[i * 4 + 1] = fminf(fmaxf(b.y, 0.f), 1024.f);
        selb[i * 4 + 2] = fminf(fmaxf(b.z, 0.f), 1024.f);
        selb[i * 4 + 3] = fminf(fmaxf(b.w, 0.f), 1024.f);
        sels[i] = scores[idx];
    }
}

// ------------------------------ greedy NMS + output ------------------------
__global__ void k_nms(const float* __restrict__ selb,
                      const float* __restrict__ sels,
                      float* __restrict__ out) {
    __shared__ float x1[PRE_K], y1[PRE_K], x2[PRE_K], y2[PRE_K], sc[PRE_K];
    __shared__ int keep[PRE_K];
    int tid = threadIdx.x;
    if (tid < PRE_K) {
        float a = selb[tid * 4 + 0], b = selb[tid * 4 + 1];
        float c = selb[tid * 4 + 2], d = selb[tid * 4 + 3];
        x1[tid] = a; y1[tid] = b; x2[tid] = c; y2[tid] = d;
        sc[tid] = sels[tid];
        keep[tid] = ((c - a) >= 0.001f) && ((d - b) >= 0.001f);
    }
    for (int p = tid; p < POST_K * 5; p += 1024) out[p] = 0.f;
    __syncthreads();

    for (int i = 0; i < PRE_K; ++i) {
        if (keep[i]) {
            if (tid > i && tid < PRE_K && keep[tid]) {
                float xx1 = fmaxf(x1[i], x1[tid]), yy1 = fmaxf(y1[i], y1[tid]);
                float xx2 = fminf(x2[i], x2[tid]), yy2 = fminf(y2[i], y2[tid]);
                float w = fmaxf(xx2 - xx1, 0.f), h = fmaxf(yy2 - yy1, 0.f);
                float inter = w * h;
                float ai = (x2[i] - x1[i]) * (y2[i] - y1[i]);
                float aj = (x2[tid] - x1[tid]) * (y2[tid] - y1[tid]);
                if (inter / (ai + aj - inter + 1e-9f) > 0.7f) keep[tid] = 0;
            }
        }
        __syncthreads();
    }

    if (tid == 0) {
        int cnt = 0;
        for (int i = 0; i < PRE_K && cnt < POST_K; ++i) {
            if (keep[i]) {
                out[cnt * 5 + 0] = x1[i];
                out[cnt * 5 + 1] = y1[i];
                out[cnt * 5 + 2] = x2[i];
                out[cnt * 5 + 3] = y2[i];
                out[cnt * 5 + 4] = sc[i];
                ++cnt;
            }
        }
    }
}

// ---------------------------------------------------------------------------
extern "C" void kernel_launch(void* const* d_in, const int* in_sizes, int n_in,
                              void* d_out, int out_size, void* d_ws, size_t ws_size,
                              hipStream_t stream) {
    (void)in_sizes; (void)n_in; (void)out_size; (void)ws_size;
    const float* fmap   = (const float*)d_in[0];
    const float* base_w = (const float*)d_in[1];
    const float* base_b = (const float*)d_in[2];
    const float* cls_w  = (const float*)d_in[3];
    const float* cls_b  = (const float*)d_in[4];
    const float* reg_w  = (const float*)d_in[5];
    const float* reg_b  = (const float*)d_in[6];
    float* out = (float*)d_out;

    size_t off = 0;
    auto alloc = [&](size_t bytes) -> void* {
        void* p = (char*)d_ws + off;
        off += (bytes + 255) & ~(size_t)255;
        return p;
    };
    unsigned short* ftp   = (unsigned short*)alloc((size_t)PADW * PADW * CCH * 2);
    unsigned short* w9    = (unsigned short*)alloc((size_t)9 * CCH * CCH * 2);
    unsigned short* hw    = (unsigned short*)alloc((size_t)48 * CCH * 2);
    unsigned short* xt    = (unsigned short*)alloc((size_t)NPIX * CCH * 2);
    float* out45          = (float*)alloc((size_t)NPIX * 48 * 4);
    float* scores         = (float*)alloc((size_t)NANCH * 4);
    float* boxes          = (float*)alloc((size_t)NANCH * 16);
    unsigned* state       = (unsigned*)alloc(64);
    unsigned* hist        = (unsigned*)alloc(256 * 4);
    unsigned* candk       = (unsigned*)alloc(CANDN * 4);
    unsigned* candi       = (unsigned*)alloc(CANDN * 4);
    float* selb           = (float*)alloc((size_t)PRE_K * 16);
    float* sels           = (float*)alloc((size_t)PRE_K * 4);

    // 1. pack weights (conv + heads) into bf16 GEMM layouts
    {
        int tot = 9 * CCH * CCH + 48 * CCH;
        k_pack_w<<<(tot + 255) / 256, 256, 0, stream>>>(base_w, cls_w, reg_w, w9, hw);
    }
    // 2. zero halo + transpose fmap -> padded [row][col][c] bf16
    k_zero_edge<<<514, 256, 0, stream>>>(ftp);
    k_tpose<<<dim3(NPIX / 32, CCH / 32), dim3(32, 8), 0, stream>>>(fmap, ftp);
    // 3. conv 3x3 + bias + relu (WMMA, branch-free)
    k_conv<<<dim3(NPIX / 512, CCH / 16), 256, 0, stream>>>(ftp, w9, base_b, xt);
    // 4. heads (WMMA)
    k_head<<<dim3(NPIX / 512, 3), 256, 0, stream>>>(xt, hw, out45);
    // 5. sigmoid + decode
    k_decode<<<NANCH / 256, 256, 0, stream>>>(out45, cls_b, reg_b, scores, boxes);
    // 6. radix select threshold for top-1000
    k_sel_init<<<1, 1, 0, stream>>>(state);
    k_zero_cand<<<CANDN / 256, 256, 0, stream>>>(candk, candi);
    const int shifts[4]      = {24, 16, 8, 0};
    const unsigned masks[4]  = {0x00000000u, 0xFF000000u, 0xFFFF0000u, 0xFFFFFF00u};
    for (int p = 0; p < 4; ++p) {
        k_zero_hist<<<1, 256, 0, stream>>>(hist);
        k_hist<<<NANCH / 256, 256, 0, stream>>>(scores, state, hist, shifts[p], masks[p]);
        k_pick<<<1, 1, 0, stream>>>(state, hist, shifts[p]);
    }
    // 7. compact candidates >= threshold, sort by (score desc, idx asc)
    k_compact<<<NANCH / 256, 256, 0, stream>>>(scores, state, candk, candi);
    k_sort<<<1, 1024, 0, stream>>>(candk, candi);
    // 8. gather + clip top-1000
    k_gather<<<(PRE_K + 255) / 256, 256, 0, stream>>>(candi, scores, boxes, selb, sels);
    // 9. greedy NMS + write 300x5 dets
    k_nms<<<1, 1024, 0, stream>>>(selb, sels, out);
}
